// ExpertsChooseMLP_25348896981198
// MI455X (gfx1250) — matrix operand order repaired and checked by reading, not compiled, not run on hardware
//
#include <hip/hip_runtime.h>
#include <hip/hip_bf16.h>

// ---------------------------------------------------------------------------
// ExpertsChooseMLP for MI455X (gfx1250, wave32, WMMA + TDM)
// B=4, T=4096, D=2048, E=8, C=1024, H=256
// bf16 WMMA (f32 accumulate); W1/W2 staged into LDS via Tensor Data Mover
// (double-buffered; fallback: load+ds_store); software-pipelined fragment
// loads; f32 atomics for the gated scatter-add combine.
// ---------------------------------------------------------------------------

typedef __attribute__((ext_vector_type(16))) __bf16 v16bf;
typedef __attribute__((ext_vector_type(8)))  __bf16 v8bf;
typedef __attribute__((ext_vector_type(8)))  float  v8f;
typedef __attribute__((ext_vector_type(4)))  float  v4f;

#define B_  4
#define T_  4096
#define D_  2048
#define E_  8
#define C_  1024
#define H_  256
#define CT_ 128          // c-tile per workgroup

// ---- Tensor Data Mover availability ---------------------------------------
#ifndef USE_TDM
#if defined(__has_builtin)
#if __has_builtin(__builtin_amdgcn_tensor_load_to_lds)
#define USE_TDM 1
#else
#define USE_TDM 0
#endif
#else
#define USE_TDM 0
#endif
#endif

__device__ __forceinline__ unsigned lds_off(const void* p) {
    // LDS aperture keeps the byte offset in the low 32 bits of a generic ptr.
    return (unsigned)(unsigned long long)p;
}

#if USE_TDM
typedef __attribute__((ext_vector_type(4))) unsigned int u32x4;
typedef __attribute__((ext_vector_type(8))) int          i32x8;
typedef __attribute__((ext_vector_type(4))) int          i32x4;

// 2D tile load Global->LDS via TDM (bf16 elements), per CDNA5 ISA.
__device__ __forceinline__ void tdm_load_2d(unsigned lds_byte_off,
                                            const void* gsrc,
                                            unsigned tile_w,      // elems (dim0)
                                            unsigned tile_h,      // rows  (dim1)
                                            unsigned stride_elems) {
    unsigned long long ga = (unsigned long long)gsrc;
    u32x4 g0;
    g0[0] = 1u;                                            // count=1 (valid D#)
    g0[1] = lds_byte_off;                                  // lds_addr
    g0[2] = (unsigned)ga;                                  // global_addr[31:0]
    g0[3] = (unsigned)((ga >> 32) & 0x1FFFFFFull)          // global_addr[56:32]
          | (2u << 30);                                    // type=2 ("image")
    i32x8 g1;
    g1[0] = (int)(1u << 16);                               // data_size=1 -> 2B
    g1[1] = (int)((tile_w & 0xFFFFu) << 16);               // tensor_dim0[15:0]
    g1[2] = (int)(((tile_w >> 16) & 0xFFFFu)               // tensor_dim0[31:16]
          | ((tile_h & 0xFFFFu) << 16));                   // tensor_dim1[15:0]
    g1[3] = (int)(((tile_h >> 16) & 0xFFFFu)               // tensor_dim1[31:16]
          | ((tile_w & 0xFFFFu) << 16));                   // tile_dim0
    g1[4] = (int)(tile_h & 0xFFFFu);                       // tile_dim1 (dim2=0)
    g1[5] = (int)stride_elems;                             // tensor_dim0_stride
    g1[6] = 0;
    g1[7] = 0;
    i32x4 z4 = {0, 0, 0, 0};
#if defined(__clang_major__) && (__clang_major__ >= 23)
    i32x8 z8 = {0, 0, 0, 0, 0, 0, 0, 0};
    __builtin_amdgcn_tensor_load_to_lds(g0, g1, z4, z4, z8, 0);
#else
    __builtin_amdgcn_tensor_load_to_lds(g0, g1, z4, z4, 0);
#endif
}
#endif // USE_TDM

// ---- float32 -> bf16 weight conversion (W1, W2) ---------------------------
__global__ void cvt_f32_to_bf16_kernel(const float* __restrict__ in,
                                       __bf16* __restrict__ out, int n) {
    int i = (blockIdx.x * blockDim.x + threadIdx.x) * 8;
    if (i >= n) return;
    v8f v = *(const v8f*)(in + i);
    v8bf o;
#pragma unroll
    for (int j = 0; j < 8; ++j) o[j] = (__bf16)v[j];
    *(v8bf*)(out + i) = o;
}

// ---- zero-init the scatter-add destination --------------------------------
__global__ void zero_out_kernel(float* __restrict__ out, int n) {
    int i = (blockIdx.x * blockDim.x + threadIdx.x) * 4;
    if (i >= n) return;
    v4f z = {0.f, 0.f, 0.f, 0.f};
    *(v4f*)(out + i) = z;
}

// ---- fused MoE MLP: gather -> GEMM1 -> GELU -> GEMM2 -> gated scatter-add --
__global__ __launch_bounds__(256) void
moe_mlp_kernel(const float*  __restrict__ x,      // (B,T,D) f32
               const __bf16* __restrict__ w1b,    // (E,H,D) bf16
               const __bf16* __restrict__ w2b,    // (E,D,H) bf16
               const float*  __restrict__ b1,     // (E,H)
               const float*  __restrict__ b2,     // (D,)
               const int*    __restrict__ idx,    // (B,E,C)
               const float*  __restrict__ gate,   // (B,E,C)
               float*        __restrict__ out) {  // (B,T,D)
    // 64KB LDS, time-multiplexed:
    //   phase 1: two 32KB W1 k-slabs (256 rows x 64 cols, double buffered)
    //   phase 2: 128x256 bf16 h tile (transpose buffer)
    //   phase 3: two 32KB W2 chunks (64 rows x 256 cols, double buffered)
    __shared__ __bf16 smem[32768];

    const int blk   = blockIdx.x;
    const int ctile = blk & 7;          // C_/CT_ = 8
    const int e     = (blk >> 3) & 7;   // E_ = 8
    const int b     = blk >> 6;         // B_ = 4

    const int tid   = threadIdx.x;
    const int lane  = tid & 31;
    const int wave  = tid >> 5;
    const int l16   = lane & 15;
    const int lhalf = lane >> 4;

    const size_t ieg_base = ((size_t)b * E_ + e) * C_ + (size_t)ctile * CT_;

    // ---- per-lane gathered-x row pointer (A-matrix rows for GEMM1) --------
    // A k order used here (16x32 bf16, B matches): lane<16 holds K 0..15; lane>=16: 16..31
    const int    rowc  = wave * 16 + l16;
    const int    tok_a = idx[ieg_base + rowc];
    const float* px    = x + ((size_t)b * T_ + tok_a) * D_ + lhalf * 16;

    const __bf16* w1e = w1b + (size_t)e * H_ * D_;
    const __bf16* w2e = w2b + (size_t)e * D_ * H_;

    // ---- GEMM1: h_strip(16x256) = sel(16x2048) @ W1[e]^T ------------------
    v8f acc[16];
#pragma unroll
    for (int i = 0; i < 16; ++i) {
        v8f z = {0.f, 0.f, 0.f, 0.f, 0.f, 0.f, 0.f, 0.f};
        acc[i] = z;
    }

    // A prefetch (k-step 0)
    v8f a0 = *(const v8f*)(px);
    v8f a1 = *(const v8f*)(px + 8);

#if USE_TDM
    {   // double-buffered TDM staging: slab = 256 rows x 64 cols (32KB)
        const int boff1 = l16 * 64 + lhalf * 16;
        if (tid == 0) tdm_load_2d(lds_off(smem), w1e, 64, H_, D_);
        for (int kso = 0; kso < 32; ++kso) {
            if (tid == 0) __builtin_amdgcn_s_wait_tensorcnt(0);
            __syncthreads();
            if (tid == 0 && kso + 1 < 32)
                tdm_load_2d(lds_off(smem + ((kso + 1) & 1) * 16384),
                            w1e + (size_t)(kso + 1) * 64, 64, H_, D_);
            const __bf16* slab = smem + (kso & 1) * 16384;
#pragma unroll
            for (int ki = 0; ki < 2; ++ki) {
                v16bf afrag;
#pragma unroll
                for (int j = 0; j < 8; ++j) {
                    afrag[j]     = (__bf16)a0[j];
                    afrag[j + 8] = (__bf16)a1[j];
                }
                // prefetch next k-step's A while WMMAs run
                const int knext = kso * 64 + ki * 32 + 32;
                if (knext < D_) {
                    a0 = *(const v8f*)(px + knext);
                    a1 = *(const v8f*)(px + knext + 8);
                }
                const __bf16* bb = slab + ki * 32 + boff1;
                v16bf bcur = *(const v16bf*)(bb);
#pragma unroll
                for (int nt = 0; nt < 16; ++nt) {
                    v16bf bnext = bcur;
                    if (nt < 15) bnext = *(const v16bf*)(bb + (nt + 1) * 1024);
                    acc[nt] = __builtin_amdgcn_wmma_f32_16x16x32_bf16(
                        false, afrag, false, bcur, (short)0, acc[nt], false, false);
                    bcur = bnext;
                }
            }
        }
    }
#else
    {   // fallback: single-buffered manual staging, 32-col slabs
        const int boff1 = l16 * 32 + lhalf * 16;
        for (int ks = 0; ks < D_ / 32; ++ks) {
            const int k0 = ks * 32;
            __syncthreads();
            {   // thread t copies W1[e] row t, cols k0..k0+31
                const __bf16* src = w1e + (size_t)tid * D_ + k0;
                __bf16*       dst = smem + tid * 32;
#pragma unroll
                for (int j = 0; j < 4; ++j)
                    *(v8bf*)(dst + j * 8) = *(const v8bf*)(src + j * 8);
            }
            __syncthreads();
            v16bf afrag;
#pragma unroll
            for (int j = 0; j < 8; ++j) {
                afrag[j]     = (__bf16)a0[j];
                afrag[j + 8] = (__bf16)a1[j];
            }
            if (k0 + 32 < D_) {
                a0 = *(const v8f*)(px + k0 + 32);
                a1 = *(const v8f*)(px + k0 + 40);
            }
            const __bf16* bb = smem + boff1;
            v16bf bcur = *(const v16bf*)(bb);
#pragma unroll
            for (int nt = 0; nt < 16; ++nt) {
                v16bf bnext = bcur;
                if (nt < 15) bnext = *(const v16bf*)(bb + (nt + 1) * 512);
                acc[nt] = __builtin_amdgcn_wmma_f32_16x16x32_bf16(
                    false, afrag, false, bcur, (short)0, acc[nt], false, false);
                bcur = bnext;
            }
        }
    }
#endif

    // ---- bias + exact GELU, transpose via LDS (bf16) ----------------------
    __syncthreads();
#pragma unroll
    for (int nt = 0; nt < 16; ++nt) {
        const float bv = b1[e * H_ + nt * 16 + l16];
#pragma unroll
        for (int r = 0; r < 8; ++r) {
            float v = acc[nt][r] + bv;
            v = 0.5f * v * (1.0f + erff(v * 0.70710678118654752f));
            const int row = wave * 16 + r + 8 * lhalf;     // C layout M index
            smem[row * 256 + nt * 16 + l16] = (__bf16)v;
        }
    }
    __syncthreads();

    // ---- preload GEMM2 A fragments from LDS (row = strip row l16) ---------
    v16bf a2[8];
    const __bf16* ph = smem + (wave * 16 + l16) * 256 + lhalf * 16;
#pragma unroll
    for (int kk = 0; kk < 8; ++kk) {
        v8bf c0 = *(const v8bf*)(ph + kk * 32);
        v8bf c1 = *(const v8bf*)(ph + kk * 32 + 8);
        a2[kk] = __builtin_shufflevector(c0, c1, 0, 1, 2, 3, 4, 5, 6, 7,
                                                 8, 9, 10, 11, 12, 13, 14, 15);
    }

    // ---- per-lane token/gate for the 8 output rows this lane touches ------
    int   tok_r[8];
    float g_r[8];
#pragma unroll
    for (int r = 0; r < 8; ++r) {
        const int m = wave * 16 + r + 8 * lhalf;
        tok_r[r] = idx[ieg_base + m];
        g_r[r]   = gate[ieg_base + m];
    }

    // ---- GEMM2: y_strip(16x2048) = h_strip(16x256) @ W2[e]^T, gated scatter
    float*    outb  = out + (size_t)b * T_ * D_;
    const int boff2 = l16 * 256 + lhalf * 16;   // row l16 (pitch 256), K half

    __syncthreads();   // all waves done reading h from smem

#if USE_TDM
    if (tid == 0) tdm_load_2d(lds_off(smem), w2e, H_, 64, H_);
    for (int cc = 0; cc < 32; ++cc) {           // 32 chunks of 64 W2 rows
        if (tid == 0) __builtin_amdgcn_s_wait_tensorcnt(0);
        __syncthreads();
        if (tid == 0 && cc + 1 < 32)
            tdm_load_2d(lds_off(smem + ((cc + 1) & 1) * 16384),
                        w2e + (size_t)(cc + 1) * 64 * H_, H_, 64, H_);
        const __bf16* chunk = smem + (cc & 1) * 16384;
#pragma unroll
        for (int ntl = 0; ntl < 4; ++ntl) {
            const int nt2 = cc * 4 + ntl;
            v8f acc2 = {0.f, 0.f, 0.f, 0.f, 0.f, 0.f, 0.f, 0.f};
            const __bf16* bbase = chunk + ntl * 4096 + boff2;
            v16bf bcur = *(const v16bf*)(bbase);
#pragma unroll
            for (int kk = 0; kk < 8; ++kk) {
                v16bf bnext = bcur;
                if (kk < 7) bnext = *(const v16bf*)(bbase + (kk + 1) * 32);
                acc2 = __builtin_amdgcn_wmma_f32_16x16x32_bf16(
                    false, a2[kk], false, bcur, (short)0, acc2, false, false);
                bcur = bnext;
            }
            const int   col = nt2 * 16 + l16;
            const float bv  = b2[col];
#pragma unroll
            for (int r = 0; r < 8; ++r) {
                const float v = (acc2[r] + bv) * g_r[r];
                unsafeAtomicAdd(outb + (size_t)tok_r[r] * D_ + col, v);
            }
        }
    }
#else
    for (int cc = 0; cc < 16; ++cc) {           // 16 chunks of 128 W2 rows
        __syncthreads();
        {   // thread t copies half of W2 row cc*128 + t/2
            const __bf16* src = w2e + ((size_t)cc * 128 + (tid >> 1)) * H_
                              + (tid & 1) * 128;
            __bf16*       dst = smem + (tid >> 1) * 256 + (tid & 1) * 128;
#pragma unroll
            for (int j = 0; j < 16; ++j)
                *(v8bf*)(dst + j * 8) = *(const v8bf*)(src + j * 8);
        }
        __syncthreads();
#pragma unroll
        for (int ntl = 0; ntl < 8; ++ntl) {
            const int nt2 = cc * 8 + ntl;
            v8f acc2 = {0.f, 0.f, 0.f, 0.f, 0.f, 0.f, 0.f, 0.f};
            const __bf16* bbase = smem + ntl * 4096 + boff2;
            v16bf bcur = *(const v16bf*)(bbase);
#pragma unroll
            for (int kk = 0; kk < 8; ++kk) {
                v16bf bnext = bcur;
                if (kk < 7) bnext = *(const v16bf*)(bbase + (kk + 1) * 32);
                acc2 = __builtin_amdgcn_wmma_f32_16x16x32_bf16(
                    false, a2[kk], false, bcur, (short)0, acc2, false, false);
                bcur = bnext;
            }
            const int   col = nt2 * 16 + l16;
            const float bv  = b2[col];
#pragma unroll
            for (int r = 0; r < 8; ++r) {
                const float v = (acc2[r] + bv) * g_r[r];
                unsafeAtomicAdd(outb + (size_t)tok_r[r] * D_ + col, v);
            }
        }
    }
#endif
}

// ---------------------------------------------------------------------------
extern "C" void kernel_launch(void* const* d_in, const int* in_sizes, int n_in,
                              void* d_out, int out_size, void* d_ws, size_t ws_size,
                              hipStream_t stream) {
    const float* x    = (const float*)d_in[0];
    const float* W1   = (const float*)d_in[1];
    const float* b1   = (const float*)d_in[2];
    const float* W2   = (const float*)d_in[3];
    const float* b2   = (const float*)d_in[4];
    const int*   idx  = (const int*)d_in[5];
    const float* gate = (const float*)d_in[6];
    float*       out  = (float*)d_out;

    // workspace: bf16 copies of W1, W2 (8 MB each)
    const int wElems = E_ * H_ * D_;            // 4,194,304
    __bf16* w1b = (__bf16*)d_ws;
    __bf16* w2b = w1b + (size_t)wElems;

    {
        int thr = 256, blkW = (wElems / 8 + thr - 1) / thr;
        cvt_f32_to_bf16_kernel<<<blkW, thr, 0, stream>>>(W1, w1b, wElems);
        cvt_f32_to_bf16_kernel<<<blkW, thr, 0, stream>>>(W2, w2b, wElems);
    }
    {
        const int n = B_ * T_ * D_;             // 33,554,432
        int thr = 256, blkZ = (n / 4 + thr - 1) / thr;
        zero_out_kernel<<<blkZ, thr, 0, stream>>>(out, n);
    }
    {
        dim3 grid(B_ * E_ * (C_ / CT_));        // 256 workgroups
        dim3 block(256);                         // 8 wave32
        moe_mlp_kernel<<<grid, block, 0, stream>>>(x, w1b, w2b, b1, b2,
                                                   idx, gate, out);
    }
}
